// deform_conv_52441550684808
// MI455X (gfx1250) — compile-verified, matrix operand order checked by reading, not run on hardware
//
#include <hip/hip_runtime.h>

typedef __attribute__((ext_vector_type(16))) __bf16 v16bf;
typedef __attribute__((ext_vector_type(8)))  float  v8f;

#define Bn   4
#define Cn   64
#define On   64
#define Hn   128
#define Wn   128
#define K2n  9
#define NOFF 36      // 2*OG*K2
#define NMSK 18      // OG*K2
#define KD   576     // C*K2 (GEMM K dim)
#define MT   4       // M-tiles of 16 rows
#define KT   18      // KD/32 K-chunks
#define HW   (Hn*Wn)
#define NFRAG (MT*KT*32*16)          // bf16 elements per (hi or lo) fragment array
#define SMEM_BYTES (2 * NFRAG * 2)   // hi + lo staged in LDS: 147456 B (320KB WGP LDS)

// K-dimension ordering for BOTH GEMMs: Kg' = t*64 + ch  (channel fastest).
// A 16x32 bf16 fragment layout (wave32): lanes 0-15 hold rows M=0..15 with
// Klocal {0..7,16..23}; lanes 16-31 same rows with Klocal {8..15,24..31}.
// Element e: Klocal = e + (e&8) + 8*half. Chunk kk: t = kk>>1, ch-block = (kk&1)*32.
// B mirrors A with N<->M (cols = pixels). C/D: col N = lane&15, VGPR r -> M = r + 8*half.

// ---------------------------------------------------------------------------
// Kernel 0: pre-swizzle weights into WMMA A-fragment order, bf16 hi+lo split.
// Array A: w_main (64 x 576). Array B: [w_off; w_mask; zeros] (64 x 576).
// Each array stored hi-block then lo-block contiguously (2*NFRAG bf16).
// ---------------------------------------------------------------------------
__global__ void dcn_prep_w(const float* __restrict__ w_main,
                           const float* __restrict__ w_off,
                           const float* __restrict__ w_mask,
                           __bf16* __restrict__ WA,    // [hi: NFRAG][lo: NFRAG]
                           __bf16* __restrict__ WB)    // [hi: NFRAG][lo: NFRAG]
{
    const int i = blockIdx.x * blockDim.x + threadIdx.x;
    if (i >= 2 * NFRAG) return;
    const bool second = i >= NFRAG;
    const int j    = second ? i - NFRAG : i;
    const int e    = j & 15;
    const int lane = (j >> 4) & 31;
    const int kk   = (j >> 9) % KT;
    const int mt   = (j >> 9) / KT;
    const int l    = lane & 15;
    const int half = lane >> 4;
    const int o    = mt * 16 + l;
    const int Kl   = e + (e & 8) + 8 * half;        // 0..31
    const int t    = kk >> 1;                       // tap 0..8
    const int ch   = (kk & 1) * 32 + Kl;            // channel 0..63

    // main weights
    {
        const float v = w_main[((size_t)o * Cn + ch) * K2n + t];
        const __bf16 hi = (__bf16)v;
        if (!second) WA[j] = hi;
        else         WA[NFRAG + j] = (__bf16)(v - (float)hi);
    }
    // offset+mask weights (rows 54..63 zero-padded)
    {
        float v;
        if (o < NOFF)             v = w_off [((size_t)o * Cn + ch) * K2n + t];
        else if (o < NOFF + NMSK) v = w_mask[((size_t)(o - NOFF) * Cn + ch) * K2n + t];
        else                      v = 0.f;
        const __bf16 hi = (__bf16)v;
        if (!second) WB[j] = hi;
        else         WB[NFRAG + j] = (__bf16)(v - (float)hi);
    }
}

// ---------------------------------------------------------------------------
// Cooperative copy of both fragment blocks (hi+lo) into LDS.
// ---------------------------------------------------------------------------
__device__ __forceinline__ void stage_frags_lds(__bf16* __restrict__ sfrag,
                                                const __bf16* __restrict__ Wfrag)
{
    const uint4* __restrict__ src = (const uint4*)Wfrag;
    uint4* __restrict__ dst = (uint4*)sfrag;
#pragma unroll
    for (int it = 0; it < SMEM_BYTES / 16 / 256; ++it)
        dst[it * 256 + threadIdx.x] = src[it * 256 + threadIdx.x];
    __syncthreads();
}

// ---------------------------------------------------------------------------
// Shared WMMA tail: 4 M-tiles x (A_hi*B_hi + A_hi*B_lo + A_lo*B_hi),
// A fragments read from LDS (ds_load_b128).
// ---------------------------------------------------------------------------
__device__ __forceinline__ void wmma_acc3(v8f acc[MT],
                                          const __bf16* __restrict__ sfrag,
                                          int kk, int lane,
                                          const v16bf& bhi, const v16bf& blo)
{
#pragma unroll
    for (int m = 0; m < MT; ++m) {
        const int fi = ((m * KT + kk) * 32 + lane) * 16;
        const v16bf ah = *(const v16bf*)(sfrag + fi);
        const v16bf al = *(const v16bf*)(sfrag + NFRAG + fi);
        acc[m] = __builtin_amdgcn_wmma_f32_16x16x32_bf16(
            false, ah, false, bhi, (short)0, acc[m], false, false);
        acc[m] = __builtin_amdgcn_wmma_f32_16x16x32_bf16(
            false, ah, false, blo, (short)0, acc[m], false, false);
        acc[m] = __builtin_amdgcn_wmma_f32_16x16x32_bf16(
            false, al, false, bhi, (short)0, acc[m], false, false);
    }
}

__device__ __forceinline__ int clampi(int v, int lo, int hi)
{
    return v < lo ? lo : (v > hi ? hi : v);
}

// ---------------------------------------------------------------------------
// Kernel 1: offset+mask conv as WMMA GEMM (M=54 pad 64, K=576, N=65536).
// im2col B: per chunk the tap (dy,dx) is uniform -> one bounds check,
// 16 channel loads at constant immediate offsets.
// ---------------------------------------------------------------------------
__global__ __launch_bounds__(256)
void dcn_offmask_wmma(const float* __restrict__ x,
                      const __bf16* __restrict__ WB,
                      const float* __restrict__ b_off,
                      const float* __restrict__ b_mask,
                      float* __restrict__ off,
                      float* __restrict__ msk)
{
    extern __shared__ __align__(16) unsigned char smem_raw[];
    __bf16* sfrag = (__bf16*)smem_raw;
    stage_frags_lds(sfrag, WB);

    const int wave = threadIdx.x >> 5;
    const int lane = threadIdx.x & 31;
    const int l    = lane & 15;
    const int half = lane >> 4;

    const int tile = blockIdx.x * 8 + wave;
    const int p0   = tile << 4;
    const int b    = p0 >> 14;
    const int h    = (p0 >> 7) & (Hn - 1);
    const int w0   = p0 & (Wn - 1);
    const int wc   = w0 + l;
    const int pix  = h * Wn + wc;
    const float* xB = x + (size_t)b * Cn * HW;

    v8f acc[MT];
#pragma unroll
    for (int m = 0; m < MT; ++m)
#pragma unroll
        for (int r = 0; r < 8; ++r) acc[m][r] = 0.f;

    for (int kk = 0; kk < KT; ++kk) {
        const int t  = kk >> 1;
        const int ki = t / 3;
        const int kj = t - ki * 3;
        const int yy = h - 1 + ki;
        const int xx = wc - 1 + kj;
        const bool ok = (yy >= 0) & (yy < Hn) & (xx >= 0) & (xx < Wn);
        const int idx = clampi(yy, 0, Hn - 1) * Wn + clampi(xx, 0, Wn - 1);
        const float* xg = xB + (size_t)((kk & 1) * 32 + 8 * half) * HW + idx;

        v16bf bhi, blo;
#pragma unroll
        for (int e = 0; e < 16; ++e) {
            const float v = ok ? xg[(size_t)(e + (e & 8)) * HW] : 0.f;
            const __bf16 vh = (__bf16)v;
            bhi[e] = vh;
            blo[e] = (__bf16)(v - (float)vh);
        }
        wmma_acc3(acc, sfrag, kk, lane, bhi, blo);
    }

    // store: rows 0..35 -> offset (+bias), rows 36..53 -> sigmoid(mask+bias)
    float* offB = off + (size_t)b * NOFF * HW + pix;
    float* mskB = msk + (size_t)b * NMSK * HW + pix;
#pragma unroll
    for (int m = 0; m < MT; ++m)
#pragma unroll
        for (int r = 0; r < 8; ++r) {
            const int o = m * 16 + half * 8 + r;
            const float v = acc[m][r];
            if (o < NOFF) {
                offB[(size_t)o * HW] = v + b_off[o];
            } else if (o < NOFF + NMSK) {
                const float z = v + b_mask[o - NOFF];
                mskB[(size_t)(o - NOFF) * HW] = 1.f / (1.f + __expf(-z));
            }
        }
}

// ---------------------------------------------------------------------------
// Kernel 2: deformable sampling + GEMM. Per chunk, (g,t) uniform:
// one offset pair + mask + floor/frac + validity-folded corner weights,
// then 16 channels x 4 corner loads (constant immediate offsets) + 3 FMA.
// ---------------------------------------------------------------------------
__global__ __launch_bounds__(256)
void dcn_main_wmma(const float* __restrict__ x,
                   const float* __restrict__ off,
                   const float* __restrict__ msk,
                   const __bf16* __restrict__ WA,
                   float* __restrict__ out)
{
    extern __shared__ __align__(16) unsigned char smem_raw[];
    __bf16* sfrag = (__bf16*)smem_raw;
    stage_frags_lds(sfrag, WA);

    const int wave = threadIdx.x >> 5;
    const int lane = threadIdx.x & 31;
    const int l    = lane & 15;
    const int half = lane >> 4;

    const int tile = blockIdx.x * 8 + wave;
    const int p0   = tile << 4;
    const int b    = p0 >> 14;
    const int h    = (p0 >> 7) & (Hn - 1);
    const int w0   = p0 & (Wn - 1);
    const int wc   = w0 + l;
    const int pix  = h * Wn + wc;

    const float* offB = off + (size_t)b * NOFF * HW;
    const float* mskB = msk + (size_t)b * NMSK * HW;
    const float* xB   = x   + (size_t)b * Cn * HW;

    v8f acc[MT];
#pragma unroll
    for (int m = 0; m < MT; ++m)
#pragma unroll
        for (int r = 0; r < 8; ++r) acc[m][r] = 0.f;

    for (int kk = 0; kk < KT; ++kk) {
        const int t  = kk >> 1;       // tap 0..8
        const int g  = kk & 1;        // offset group
        const int ki = t / 3;
        const int kj = t - ki * 3;

        // prefetch next chunk's offset line (global_prefetch_b8)
        if (kk + 1 < KT) {
            const int tn = (kk + 1) >> 1, gn = (kk + 1) & 1;
            __builtin_prefetch(offB + (size_t)((gn * K2n + tn) * 2) * HW + pix, 0, 0);
        }

        const float offy = offB[(size_t)((g * K2n + t) * 2 + 0) * HW + pix];
        const float offx = offB[(size_t)((g * K2n + t) * 2 + 1) * HW + pix];
        const float mval = mskB[(size_t)(g * K2n + t) * HW + pix];

        const float py = offy + (float)(h - 1 + ki);
        const float px = offx + (float)(wc - 1 + kj);
        const float y0f = floorf(py);
        const float x0f = floorf(px);
        const float wy = py - y0f;
        const float wx = px - x0f;
        const int y0 = (int)y0f, x0 = (int)x0f;
        const int y1 = y0 + 1,   x1 = x0 + 1;

        const bool vy0 = (y0 >= 0) & (y0 < Hn);
        const bool vy1 = (y1 >= 0) & (y1 < Hn);
        const bool vx0 = (x0 >= 0) & (x0 < Wn);
        const bool vx1 = (x1 >= 0) & (x1 < Wn);

        // validity + mask folded into the 4 corner weights
        const float w00 = (vy0 & vx0) ? (1.f - wy) * (1.f - wx) * mval : 0.f;
        const float w01 = (vy0 & vx1) ? (1.f - wy) * wx * mval         : 0.f;
        const float w10 = (vy1 & vx0) ? wy * (1.f - wx) * mval         : 0.f;
        const float w11 = (vy1 & vx1) ? wy * wx * mval                 : 0.f;

        const int y0c = clampi(y0, 0, Hn - 1), y1c = clampi(y1, 0, Hn - 1);
        const int x0c = clampi(x0, 0, Wn - 1), x1c = clampi(x1, 0, Wn - 1);
        const float* xg  = xB + (size_t)(g * 32 + 8 * half) * HW;
        const float* p00 = xg + y0c * Wn + x0c;
        const float* p01 = xg + y0c * Wn + x1c;
        const float* p10 = xg + y1c * Wn + x0c;
        const float* p11 = xg + y1c * Wn + x1c;

        v16bf bhi, blo;
#pragma unroll
        for (int e = 0; e < 16; ++e) {
            const size_t co = (size_t)(e + (e & 8)) * HW;   // channel stride, imm offset
            const float vv = fmaf(w11, p11[co],
                             fmaf(w10, p10[co],
                             fmaf(w01, p01[co], w00 * p00[co])));
            const __bf16 vh = (__bf16)vv;
            bhi[e] = vh;
            blo[e] = (__bf16)(vv - (float)vh);
        }
        wmma_acc3(acc, sfrag, kk, lane, bhi, blo);
    }

    float* outB = out + (size_t)b * On * HW + pix;
#pragma unroll
    for (int m = 0; m < MT; ++m)
#pragma unroll
        for (int r = 0; r < 8; ++r) {
            const int o = m * 16 + half * 8 + r;
            outB[(size_t)o * HW] = acc[m][r];
        }
}

// ---------------------------------------------------------------------------
extern "C" void kernel_launch(void* const* d_in, const int* in_sizes, int n_in,
                              void* d_out, int out_size, void* d_ws, size_t ws_size,
                              hipStream_t stream)
{
    const float* x      = (const float*)d_in[0];
    const float* w_main = (const float*)d_in[1];
    const float* w_off  = (const float*)d_in[2];
    const float* b_off  = (const float*)d_in[3];
    const float* w_mask = (const float*)d_in[4];
    const float* b_mask = (const float*)d_in[5];
    float* out = (float*)d_out;

    // workspace layout
    float*  OFF = (float*)d_ws;                             // B*36*H*W f32
    float*  MSK = OFF + (size_t)Bn * NOFF * HW;             // B*18*H*W f32
    __bf16* WA  = (__bf16*)(MSK + (size_t)Bn * NMSK * HW);  // main weights  [hi|lo]
    __bf16* WB  = WA + 2 * (size_t)NFRAG;                   // off+mask wts  [hi|lo]

    dcn_prep_w<<<(2 * NFRAG + 255) / 256, 256, 0, stream>>>(
        w_main, w_off, w_mask, WA, WB);
    dcn_offmask_wmma<<<(Bn * HW / 16) / 8, 256, SMEM_BYTES, stream>>>(
        x, WB, b_off, b_mask, OFF, MSK);
    dcn_main_wmma<<<(Bn * HW / 16) / 8, 256, SMEM_BYTES, stream>>>(
        x, OFF, MSK, WA, out);
}
